// sequence_embedding_8564164788694
// MI455X (gfx1250) — matrix operand order repaired and checked by reading, not compiled, not run on hardware
//
#include <hip/hip_runtime.h>

typedef __attribute__((ext_vector_type(16))) __bf16 v16bf;
typedef __attribute__((ext_vector_type(8)))  __bf16 v8bf;
typedef __attribute__((ext_vector_type(4)))  __bf16 v4bf;
typedef __attribute__((ext_vector_type(8)))  float  v8f;
typedef __attribute__((ext_vector_type(4)))  float  v4f;
typedef __attribute__((ext_vector_type(4)))  int    v4i;

#define DHEAD   128
#define M_TILE  64      // Q rows per workgroup (4 waves x 16)
#define N_TILE  64      // K/V rows per iteration
#define NWAVES  4
#define BLOCK   128

#define AS1 __attribute__((address_space(1)))
#define AS3 __attribute__((address_space(3)))

// sqrt(128) * log2(e): softmax computed base-2, scale folded into Q.
#define QSCALE (11.313708498984761f * 1.4426950408889634f)

__device__ __forceinline__ v16bf cat8(v8bf lo, v8bf hi) {
  return __builtin_shufflevector(lo, hi, 0,1,2,3,4,5,6,7,8,9,10,11,12,13,14,15);
}

// ---- async global -> LDS copy (16B per lane), ASYNCcnt-tracked ----
__device__ __forceinline__ void async_b128(const void* g, void* l) {
#if __has_builtin(__builtin_amdgcn_global_load_async_to_lds_b128)
  __builtin_amdgcn_global_load_async_to_lds_b128(
      (AS1 v4i*)const_cast<void*>(g), (AS3 v4i*)l, 0, 0);
#else
  unsigned lds = (unsigned)(__UINTPTR_TYPE__)(AS3 void*)l;
  asm volatile("global_load_async_to_lds_b128 %0, %1, off"
               :: "v"(lds), "v"(g) : "memory");
#endif
}

__device__ __forceinline__ void wait_async0() {
#if __has_builtin(__builtin_amdgcn_s_wait_asynccnt)
  __builtin_amdgcn_s_wait_asynccnt(0);
#else
  asm volatile("s_wait_asynccnt 0x0" ::: "memory");
#endif
}

// =====================================================================
// Prep: p1 -> qb (bf16, pre-scaled), p2 -> kb (bf16 row-major) and
//       vt (bf16 transposed, [d][n2]). Runs once; ~14 MB of traffic.
// =====================================================================
__launch_bounds__(BLOCK)
__global__ void attn_prep(const float* __restrict__ p1,
                          const float* __restrict__ p2,
                          __bf16* __restrict__ qb,
                          __bf16* __restrict__ kb,
                          __bf16* __restrict__ vt,
                          int n1, int n2)
{
  __shared__ alignas(16) __bf16 tile[64 * DHEAD];
  const int tid = threadIdx.x;
  const int r0  = blockIdx.x * 64;
  const int r   = tid >> 1;             // 2 threads per row
  const int d0  = (tid & 1) * 64;

  if (r0 + r < n2) {
    const v4f* src = (const v4f*)(p2 + (size_t)(r0 + r) * DHEAD + d0);
    #pragma unroll
    for (int c = 0; c < 16; ++c) {
      v4f v = src[c];
      v4bf b = { (__bf16)v[0], (__bf16)v[1], (__bf16)v[2], (__bf16)v[3] };
      *(v4bf*)(tile + r * DHEAD + d0 + c * 4) = b;
      *(v4bf*)(kb + (size_t)(r0 + r) * DHEAD + d0 + c * 4) = b;
    }
  }
  if (r0 + r < n1) {
    const v4f* qs = (const v4f*)(p1 + (size_t)(r0 + r) * DHEAD + d0);
    #pragma unroll
    for (int c = 0; c < 16; ++c) {
      v4f v = qs[c];
      v4bf b = { (__bf16)(v[0] * QSCALE), (__bf16)(v[1] * QSCALE),
                 (__bf16)(v[2] * QSCALE), (__bf16)(v[3] * QSCALE) };
      *(v4bf*)(qb + (size_t)(r0 + r) * DHEAD + d0 + c * 4) = b;
    }
  }
  __syncthreads();
  if (r0 < n2) {
    const int d = tid;                  // one column per thread
    __bf16 col[64];
    #pragma unroll
    for (int rr = 0; rr < 64; ++rr) col[rr] = tile[rr * DHEAD + d];
    v8bf* dst = (v8bf*)(vt + (size_t)d * n2 + r0);
    #pragma unroll
    for (int c = 0; c < 8; ++c) {
      v8bf o;
      #pragma unroll
      for (int i = 0; i < 8; ++i) o[i] = col[c * 8 + i];
      dst[c] = o;
    }
  }
}

// =====================================================================
// Flash attention, bf16 WMMA, async double-buffered KV tiles in LDS.
// =====================================================================
__launch_bounds__(BLOCK)
__global__ void flash_attn_bf16wmma(const __bf16* __restrict__ qb,
                                    const __bf16* __restrict__ kb,
                                    const __bf16* __restrict__ vt,
                                    float* __restrict__ out,
                                    int n2)
{
  __shared__ alignas(16) __bf16 Tk[2][N_TILE * DHEAD];   // row-major K tile
  __shared__ alignas(16) __bf16 Tv[2][DHEAD * N_TILE];   // transposed V tile
  __shared__ alignas(16) __bf16 Pl[NWAVES * 16 * N_TILE];// wave-private P

  const int tid  = threadIdx.x;
  const int lane = tid & 31;
  const int wave = tid >> 5;
  const int grp  = lane >> 4;
  const int ln16 = lane & 15;
  const int qrow0 = blockIdx.x * M_TILE + wave * 16;

  // issue one KV tile (32 KB) as 16 async b128 ops per thread
  auto issue_tile = [&](int j, int buf) {
    const char* gk = (const char*)(kb + (size_t)j * DHEAD);  // contiguous 16KB
    char* lk = (char*)&Tk[buf][0];
    #pragma unroll
    for (int i = 0; i < 8; ++i) {
      const int off = (tid + i * BLOCK) * 16;
      async_b128(gk + off, lk + off);
    }
    const char* gv = (const char*)(vt + (size_t)tid * n2 + j); // 128B per row
    char* lv = (char*)&Tv[buf][tid * N_TILE];
    #pragma unroll
    for (int c = 0; c < 8; ++c)
      async_b128(gv + c * 16, lv + c * 16);
  };

  // ---- Q fragments (A-layout), already bf16 + pre-scaled ----
  v16bf qa[4];
  {
    const __bf16* qrow = qb + (size_t)(qrow0 + ln16) * DHEAD;
    #pragma unroll
    for (int kf = 0; kf < 4; ++kf) {
      v8bf lo = *(const v8bf*)(qrow + kf * 32 + grp * 8);
      v8bf hi = *(const v8bf*)(qrow + kf * 32 + 16 + grp * 8);
      qa[kf] = cat8(lo, hi);
    }
  }

  v8f o[8] = {};
  float mrow[8], lrow[8];
  #pragma unroll
  for (int i = 0; i < 8; ++i) { mrow[i] = -3.0e38f; lrow[i] = 0.0f; }

  issue_tile(0, 0);                      // prologue

  for (int j = 0; j < n2; j += N_TILE) {
    const int buf = (j / N_TILE) & 1;
    wait_async0();                       // tile j landed (per-wave)
    __syncthreads();                     // all waves: tile j ready, prev buf free
    if (j + N_TILE < n2) issue_tile(j + N_TILE, buf ^ 1); // overlap with compute

    // ---- S = (scaled Q) * K^T ----
    v8f s[4];
    #pragma unroll
    for (int nf = 0; nf < 4; ++nf) {
      v8f c = {};
      const __bf16* krow = &Tk[buf][(nf * 16 + ln16) * DHEAD];
      #pragma unroll
      for (int kf = 0; kf < 4; ++kf) {
        const v8bf* bp = (const v8bf*)(krow + kf * 32 + grp * 8);
        v16bf b = cat8(bp[0], bp[2]);
        c = __builtin_amdgcn_wmma_f32_16x16x32_bf16(
                false, qa[kf], false, b, (short)0, c, false, false);
      }
      s[nf] = c;
    }

    // ---- online softmax (base-2) ----
    float mt[8];
    #pragma unroll
    for (int i = 0; i < 8; ++i) {
      float v = s[0][i];
      v = fmaxf(v, s[1][i]); v = fmaxf(v, s[2][i]); v = fmaxf(v, s[3][i]);
      mt[i] = v;
    }
    #pragma unroll
    for (int off = 1; off < 16; off <<= 1)
      #pragma unroll
      for (int i = 0; i < 8; ++i)
        mt[i] = fmaxf(mt[i], __shfl_xor(mt[i], off, 32));

    float alpha[8];
    #pragma unroll
    for (int i = 0; i < 8; ++i) {
      float mn = fmaxf(mrow[i], mt[i]);
      alpha[i] = exp2f(mrow[i] - mn);
      mrow[i]  = mn;
    }

    float rs[8];
    #pragma unroll
    for (int i = 0; i < 8; ++i) rs[i] = 0.0f;
    #pragma unroll
    for (int nf = 0; nf < 4; ++nf)
      #pragma unroll
      for (int i = 0; i < 8; ++i) {
        float p = exp2f(s[nf][i] - mrow[i]);
        s[nf][i] = p;
        rs[i] += p;
      }
    #pragma unroll
    for (int off = 1; off < 16; off <<= 1)
      #pragma unroll
      for (int i = 0; i < 8; ++i)
        rs[i] += __shfl_xor(rs[i], off, 32);
    #pragma unroll
    for (int i = 0; i < 8; ++i) lrow[i] = lrow[i] * alpha[i] + rs[i];
    #pragma unroll
    for (int nf = 0; nf < 8; ++nf)
      #pragma unroll
      for (int i = 0; i < 8; ++i) o[nf][i] *= alpha[i];

    // ---- P: C-layout -> row-major bf16 (wave-private LDS) ----
    __bf16* pw = Pl + wave * (16 * N_TILE);
    #pragma unroll
    for (int nf = 0; nf < 4; ++nf) {
      const int n = nf * 16 + ln16;
      #pragma unroll
      for (int i = 0; i < 8; ++i)
        pw[(i + grp * 8) * N_TILE + n] = (__bf16)s[nf][i];
    }

    v16bf pa[2];
    #pragma unroll
    for (int kf = 0; kf < 2; ++kf) {
      const __bf16* prow = pw + ln16 * N_TILE;
      const v8bf* l0 = (const v8bf*)(prow + kf * 32 + grp * 8);
      const v8bf* l1 = (const v8bf*)(prow + kf * 32 + 16 + grp * 8);
      pa[kf] = cat8(*l0, *l1);
    }

    // ---- O += P * V ----
    #pragma unroll
    for (int nf = 0; nf < 8; ++nf) {
      const __bf16* vcol = &Tv[buf][(nf * 16 + ln16) * N_TILE];
      #pragma unroll
      for (int kf = 0; kf < 2; ++kf) {
        const v8bf* bp = (const v8bf*)(vcol + kf * 32 + grp * 8);
        v16bf b = cat8(bp[0], bp[2]);
        o[nf] = __builtin_amdgcn_wmma_f32_16x16x32_bf16(
                    false, pa[kf], false, b, (short)0, o[nf], false, false);
      }
    }
  }

  // ---- epilogue ----
  float inv[8];
  #pragma unroll
  for (int i = 0; i < 8; ++i) inv[i] = 1.0f / lrow[i];
  #pragma unroll
  for (int nf = 0; nf < 8; ++nf) {
    const int col = nf * 16 + ln16;
    #pragma unroll
    for (int i = 0; i < 8; ++i) {
      const int row = qrow0 + i + grp * 8;
      out[(size_t)row * DHEAD + col] = o[nf][i] * inv[i];
    }
  }
}

extern "C" void kernel_launch(void* const* d_in, const int* in_sizes, int n_in,
                              void* d_out, int out_size, void* d_ws, size_t ws_size,
                              hipStream_t stream) {
  const float* p1 = (const float*)d_in[0];
  const float* p2 = (const float*)d_in[1];
  float* out = (float*)d_out;
  const int n1 = in_sizes[0] / DHEAD;   // 8192
  const int n2 = in_sizes[1] / DHEAD;   // 8192

  __bf16* qb = (__bf16*)d_ws;                       // n1*128 bf16 (2 MB)
  __bf16* kb = qb + (size_t)n1 * DHEAD;             // n2*128 bf16 (2 MB)
  __bf16* vt = kb + (size_t)n2 * DHEAD;             // 128*n2 bf16 (2 MB)

  const int nmax = (n1 > n2 ? n1 : n2);
  attn_prep<<<dim3(nmax / 64), dim3(BLOCK), 0, stream>>>(p1, p2, qb, kb, vt, n1, n2);
  flash_attn_bf16wmma<<<dim3(n1 / M_TILE), dim3(BLOCK), 0, stream>>>(qb, kb, vt, out, n2);
}
